// GNNClassifier_58471684768359
// MI455X (gfx1250) — matrix operand order repaired and checked by reading, not compiled, not run on hardware
//
#include <hip/hip_runtime.h>

#define D       256
#define NTASK   128
#define NGRAPH  128

typedef __attribute__((ext_vector_type(16))) __bf16 v16bf;
typedef __attribute__((ext_vector_type(8)))  float  v8f;

union FragBF16 { v16bf v; unsigned int u[8]; };

__device__ __forceinline__ unsigned short f2bf(float f) {
  unsigned int u = __float_as_uint(f);
  u += 0x7FFFu + ((u >> 16) & 1u);            // round-to-nearest-even
  return (unsigned short)(u >> 16);
}
__device__ __forceinline__ float bf2f(unsigned short h) {
  return __uint_as_float((unsigned int)h << 16);
}

// ---------------------------------------------------------------- degree / isd
__global__ void deg_init(int* __restrict__ deg, int n) {
  int i = blockIdx.x * 256 + threadIdx.x;
  if (i < n) deg[i] = 1;                      // self-loop
}
__global__ void deg_accum(const int* __restrict__ dst, int* __restrict__ deg, int nE) {
  int e = blockIdx.x * 256 + threadIdx.x;
  if (e < nE) atomicAdd(deg + dst[e], 1);
}
__global__ void inv_sqrt_deg_k(const int* __restrict__ deg, float* __restrict__ isd, int n) {
  int i = blockIdx.x * 256 + threadIdx.x;
  if (i < n) isd[i] = rsqrtf((float)deg[i]);
}

// ------------------------------------------------- CSR build (one-time per call)
__device__ __forceinline__ int wave_incl_scan(int v, int lane) {
#pragma unroll
  for (int d = 1; d < 32; d <<= 1) {
    int t = __shfl_up(v, d, 32);
    if (lane >= d) v += t;
  }
  return v;
}

// exclusive scan of (deg[i]-1): single block, 256 threads, serial over tiles
__global__ void scan_offsets(const int* __restrict__ deg, int* __restrict__ offs,
                             int* __restrict__ cursor, int n) {
  __shared__ int wsum[8];
  __shared__ int carry;
  const int lane = threadIdx.x & 31, wv = threadIdx.x >> 5;
  if (threadIdx.x == 0) carry = 0;
  __syncthreads();
  for (int t0 = 0; t0 < n; t0 += 256) {
    const int i = t0 + threadIdx.x;
    const int v = (i < n) ? (deg[i] - 1) : 0;       // in-edge count (no self-loop)
    int s = wave_incl_scan(v, lane);
    if (lane == 31) wsum[wv] = s;
    __syncthreads();
    if (wv == 0 && lane < 8) {
      int ws = wsum[lane];
#pragma unroll
      for (int d2 = 1; d2 < 8; d2 <<= 1) {
        int t = __shfl_up(ws, d2, 32);
        if (lane >= d2) ws += t;
      }
      wsum[lane] = ws;                              // inclusive over waves
    }
    __syncthreads();
    const int wbase = (wv == 0) ? 0 : wsum[wv - 1];
    const int excl = carry + wbase + (s - v);
    if (i < n) { offs[i] = excl; cursor[i] = excl; }
    __syncthreads();
    if (threadIdx.x == 0) carry += wsum[7];
    __syncthreads();
  }
  if (threadIdx.x == 0) offs[n] = carry;            // total edge count
}

__global__ void csr_fill(const int* __restrict__ src, const int* __restrict__ dst,
                         int* __restrict__ cursor, int* __restrict__ csr, int nE) {
  int e = blockIdx.x * 256 + threadIdx.x;
  if (e < nE) {
    int pos = atomicAdd(cursor + dst[e], 1);
    csr[pos] = src[e];
  }
}

// --------------------------------------------- fp32 -> split bf16 (hi + lo)
__global__ void convert_x_split(const float* __restrict__ x,
                                unsigned short* __restrict__ xhi,
                                unsigned short* __restrict__ xlo, int n) {
  int i = blockIdx.x * 256 + threadIdx.x;
  if (i >= n) return;
  float f = x[i];
  unsigned short h = f2bf(f);
  xhi[i] = h;
  xlo[i] = f2bf(f - bf2f(h));
}

// W[k][n] (KxN fp32, row-major) -> Wt_{hi,lo}[n][k] bf16 (K-contiguous B frags)
__global__ void convert_wt_split(const float* __restrict__ W,
                                 unsigned short* __restrict__ Whi,
                                 unsigned short* __restrict__ Wlo, int K, int N) {
  int idx = blockIdx.x * 256 + threadIdx.x;
  if (idx >= K * N) return;
  int n = idx / K, k = idx % K;
  float f = W[(size_t)k * N + n];
  unsigned short h = f2bf(f);
  Whi[idx] = h;
  Wlo[idx] = f2bf(f - bf2f(h));
}

// ---------------------------------------------------------------- WMMA GEMM
// hs = (A @ Wt^T) * isd[row], A/W given as bf16 hi/lo splits:
// A*B ~= Ah*Bh + Al*Bh + Ah*Bl  (fp32 accumulate, ~16 effective mantissa bits)
__global__ void gemm_bf16x2_wmma(const unsigned short* __restrict__ Ahi,
                                 const unsigned short* __restrict__ Alo,
                                 const unsigned short* __restrict__ Bhi,
                                 const unsigned short* __restrict__ Blo,
                                 const float* __restrict__ isd,
                                 float* __restrict__ hs, int tilesTotal) {
  const int lane = threadIdx.x & 31;
  const int wave = threadIdx.x >> 5;
  const int t    = blockIdx.x * 8 + wave;
  if (t >= tilesTotal) return;                // wave-uniform
  const int tileN = t & 15;                   // D/16 == 16
  const int tileM = t >> 4;
  const int grp = lane >> 4;
  const int mn  = lane & 15;

  const size_t aoff = (size_t)(tileM * 16 + mn) * D;
  const size_t boff = (size_t)(tileN * 16 + mn) * D;

  v8f c = {};
#pragma unroll
  for (int k0 = 0; k0 < D; k0 += 32) {
    FragBF16 ah, al, bh, bl;
#pragma unroll
    for (int v = 0; v < 8; ++v) {
      // A 16x32 bf16: lanes0-15 K=0..15, lanes16-31 K=16..31, 2 per VGPR
      const int ka = grp * 16 + 2 * v;
      ah.u[v] = *(const unsigned int*)(Ahi + aoff + k0 + ka);
      al.u[v] = *(const unsigned int*)(Alo + aoff + k0 + ka);
      // B 32x16 bf16: lanes0-15 K=0..15, lanes16-31 K=16..31, 2 per VGPR
      const int kb = grp * 16 + 2 * v;
      bh.u[v] = *(const unsigned int*)(Bhi + boff + k0 + kb);
      bl.u[v] = *(const unsigned int*)(Blo + boff + k0 + kb);
    }
    c = __builtin_amdgcn_wmma_f32_16x16x32_bf16(false, ah.v, false, bh.v, (short)0, c, false, false);
    c = __builtin_amdgcn_wmma_f32_16x16x32_bf16(false, al.v, false, bh.v, (short)0, c, false, false);
    c = __builtin_amdgcn_wmma_f32_16x16x32_bf16(false, ah.v, false, bl.v, (short)0, c, false, false);
  }

#pragma unroll
  for (int v = 0; v < 8; ++v) {               // C/D: VGPR v -> M = v + 8*grp, N = mn
    const int m = tileM * 16 + v + grp * 8;
    const int n = tileN * 16 + mn;
    hs[(size_t)m * D + n] = c[v] * isd[m];    // pre-scale by isd[src]
  }
}

// ----------------------- CSR gather + finalize (atomic-free, L2-resident reads)
// thread = (node, float4 chunk): out = relu((hs[node] + sum_in hs[src]) * isd[node] + b)
__global__ void aggregate_finalize(const float* __restrict__ hs,
                                   const int* __restrict__ offs,
                                   const int* __restrict__ csr,
                                   const float* __restrict__ isd,
                                   const float* __restrict__ bias,
                                   unsigned short* __restrict__ xhi,
                                   unsigned short* __restrict__ xlo,
                                   float* __restrict__ hout, int writeF32, int n) {
  const int tid = blockIdx.x * 256 + threadIdx.x;
  const int node = tid >> 6, ch = tid & 63;
  if (node >= n) return;
  const size_t rb = (size_t)node * D + ch * 4;
  float4 s = *(const float4*)(hs + rb);               // self-loop term
  const int e0 = offs[node], e1 = offs[node + 1];
  for (int e = e0; e < e1; ++e) {
    const float4 v = *(const float4*)(hs + (size_t)csr[e] * D + ch * 4);
    s.x += v.x; s.y += v.y; s.z += v.z; s.w += v.w;
  }
  const float sc = isd[node];
  const float4 b4 = *(const float4*)(bias + ch * 4);
  float r[4];
  r[0] = fmaxf(s.x * sc + b4.x, 0.0f);
  r[1] = fmaxf(s.y * sc + b4.y, 0.0f);
  r[2] = fmaxf(s.z * sc + b4.z, 0.0f);
  r[3] = fmaxf(s.w * sc + b4.w, 0.0f);
#pragma unroll
  for (int j = 0; j < 4; ++j) {
    unsigned short h = f2bf(r[j]);
    xhi[rb + j] = h;
    xlo[rb + j] = f2bf(r[j] - bf2f(h));
  }
  if (writeF32) *(float4*)(hout + rb) = make_float4(r[0], r[1], r[2], r[3]);
}

// ---------------------------------------------------------------- pooling + head
// batch is sorted: graph g owns nodes [gstart[g], gstart[g+1])
__global__ void graph_starts(const int* __restrict__ batch, int* __restrict__ gstart, int n) {
  int g = blockIdx.x * 256 + threadIdx.x;
  if (g > NGRAPH) return;
  int lo = 0, hi = n;                         // lower_bound(batch, g)
  while (lo < hi) { int mid = (lo + hi) >> 1; if (batch[mid] < g) lo = mid + 1; else hi = mid; }
  gstart[g] = lo;
}

__global__ void pool_mean(const float* __restrict__ h, const int* __restrict__ gstart,
                          float* __restrict__ pooled) {
  const int tid = blockIdx.x * 256 + threadIdx.x;     // NGRAPH * 64
  const int g = tid >> 6, ch = tid & 63;
  if (g >= NGRAPH) return;
  const int i0 = gstart[g], i1 = gstart[g + 1];
  float4 s = make_float4(0.f, 0.f, 0.f, 0.f);
  for (int i = i0; i < i1; ++i) {
    const float4 v = *(const float4*)(h + (size_t)i * D + ch * 4);
    s.x += v.x; s.y += v.y; s.z += v.z; s.w += v.w;
  }
  const float inv = (i1 > i0) ? 1.0f / (float)(i1 - i0) : 0.0f;
  *(float4*)(pooled + (size_t)g * D + ch * 4) =
      make_float4(s.x * inv, s.y * inv, s.z * inv, s.w * inv);
}

__global__ void head_linear(const float* __restrict__ pooled,
                            const float* __restrict__ lin_w,
                            const float* __restrict__ lin_b,
                            float* __restrict__ out) {
  const int idx = blockIdx.x * 256 + threadIdx.x;     // NGRAPH*NTASK = 16384
  const int g = idx >> 7, t = idx & 127;
  const float* p = pooled + (size_t)g * D;
  float s = 0.0f;
#pragma unroll 4
  for (int d = 0; d < D; ++d) s += p[d] * lin_w[(size_t)d * NTASK + t];
  out[idx] = s + lin_b[t];
}

// ---------------------------------------------------------------- launcher
extern "C" void kernel_launch(void* const* d_in, const int* in_sizes, int n_in,
                              void* d_out, int out_size, void* d_ws, size_t ws_size,
                              hipStream_t stream) {
  const float* x     = (const float*)d_in[0];
  const int*   ei    = (const int*)d_in[1];
  const int*   batch = (const int*)d_in[2];
  const float* W1 = (const float*)d_in[3];  const float* b1 = (const float*)d_in[4];
  const float* W2 = (const float*)d_in[5];  const float* b2 = (const float*)d_in[6];
  const float* W3 = (const float*)d_in[7];  const float* b3 = (const float*)d_in[8];
  const float* lin_w = (const float*)d_in[9];
  const float* lin_b = (const float*)d_in[10];
  float* out = (float*)d_out;

  const int nNodes = in_sizes[0] / D;       // 10000
  const int nEdges = in_sizes[1] / 2;       // 320000
  const int* src = ei;
  const int* dst = ei + nEdges;

  // workspace layout (256B aligned)
  char* ws = (char*)d_ws;
  size_t off = 0;
  auto take = [&](size_t bytes) { char* p = ws + off; off = (off + bytes + 255) & ~(size_t)255; return p; };
  unsigned short* xhi = (unsigned short*)take((size_t)nNodes * D * 2);
  unsigned short* xlo = (unsigned short*)take((size_t)nNodes * D * 2);
  unsigned short* Whi[3], *Wlo[3];
  for (int l = 0; l < 3; ++l) {
    Whi[l] = (unsigned short*)take((size_t)D * D * 2);
    Wlo[l] = (unsigned short*)take((size_t)D * D * 2);
  }
  float* hs     = (float*)take((size_t)nNodes * D * 4);
  float* hf     = (float*)take((size_t)nNodes * D * 4);
  int*   deg    = (int*)take((size_t)nNodes * 4);
  float* isd    = (float*)take((size_t)nNodes * 4);
  int*   offs   = (int*)take((size_t)(nNodes + 1) * 4);
  int*   cursor = (int*)take((size_t)nNodes * 4);
  int*   csr    = (int*)take((size_t)nEdges * 4);
  int*   gstart = (int*)take((size_t)(NGRAPH + 1) * 4);
  float* pooled = (float*)take((size_t)NGRAPH * D * 4);
  (void)ws_size; (void)n_in; (void)out_size;

  auto cdiv = [](long long a, long long b) { return (int)((a + b - 1) / b); };

  // degrees, isd, CSR (one-time per call)
  deg_init<<<cdiv(nNodes, 256), 256, 0, stream>>>(deg, nNodes);
  deg_accum<<<cdiv(nEdges, 256), 256, 0, stream>>>(dst, deg, nEdges);
  inv_sqrt_deg_k<<<cdiv(nNodes, 256), 256, 0, stream>>>(deg, isd, nNodes);
  scan_offsets<<<1, 256, 0, stream>>>(deg, offs, cursor, nNodes);
  csr_fill<<<cdiv(nEdges, 256), 256, 0, stream>>>(src, dst, cursor, csr, nEdges);

  // split-bf16 conversions
  const float* Wsrc[3] = {W1, W2, W3};
  for (int l = 0; l < 3; ++l)
    convert_wt_split<<<cdiv(D * D, 256), 256, 0, stream>>>(Wsrc[l], Whi[l], Wlo[l], D, D);
  convert_x_split<<<cdiv((long long)nNodes * D, 256), 256, 0, stream>>>(x, xhi, xlo, nNodes * D);

  const int tilesTotal = (nNodes / 16) * (D / 16);    // 10000
  const int gemmBlocks = cdiv(tilesTotal, 8);
  const int aggBlocks  = cdiv((long long)nNodes * 64, 256);

  const float* bias[3] = {b1, b2, b3};
  for (int layer = 0; layer < 3; ++layer) {
    gemm_bf16x2_wmma<<<gemmBlocks, 256, 0, stream>>>(xhi, xlo, Whi[layer], Wlo[layer],
                                                     isd, hs, tilesTotal);
    aggregate_finalize<<<aggBlocks, 256, 0, stream>>>(hs, offs, csr, isd, bias[layer],
                                                      xhi, xlo, hf,
                                                      layer == 2 ? 1 : 0, nNodes);
  }

  // sorted-batch mean pool + head (atomic-free, deterministic)
  graph_starts<<<1, 256, 0, stream>>>(batch, gstart, nNodes);
  pool_mean<<<cdiv((long long)NGRAPH * 64, 256), 256, 0, stream>>>(hf, gstart, pooled);
  head_linear<<<cdiv(NGRAPH * NTASK, 256), 256, 0, stream>>>(pooled, lin_w, lin_b, out);
}
